// LSTMEDModule_25331717112366
// MI455X (gfx1250) — matrix operand order repaired and checked by reading, not compiled, not run on hardware
//
#include <hip/hip_runtime.h>

// LSTM encoder-decoder for MI455X (gfx1250), wave32 + WMMA f16 16x16x32.
//
// Latency-bound recurrence strategy:
//  * batch-parallel: 8 workgroups x 16 batch rows, 16 waves each own 32 z-cols
//  * weights pinned in VGPRs as pre-swizzled f16 WMMA B-fragments
//  * encoder x @ Wih^T hoisted into a fully parallel pre-GEMM (whole chip),
//    recurrent step seeds WMMA accumulators from the precomputed zx;
//    next step's zx tile is prefetched (global_prefetch) a step ahead
//  * decoder feedback folded into the recurrent weight:
//      z = h @ (Wo^T Wih^T + Whh^T) + (bo Wih^T + b)  -> one K=128 matmul/step
//  * 2 barriers per step; activations on the transcendental unit
//    (v_tanh_f32 / v_exp_f32 / v_rcp_f32), no libm branches
//  * streaming data (zx, out) uses nontemporal loads/stores

typedef __attribute__((ext_vector_type(16))) _Float16 v16h;
typedef __attribute__((ext_vector_type(8)))  float    v8f;

#define B_   128
#define T_   2048
#define F_   64
#define H_   128
#define G4H_ 512
#define NT_  512   // 16 waves

static constexpr float LOG2E_ = 1.4426950408889634f;

__device__ __forceinline__ float fast_sigmoid(float x) {
  float e = __builtin_amdgcn_exp2f(-LOG2E_ * x);       // v_exp_f32
  return __builtin_amdgcn_rcpf(1.0f + e);              // v_rcp_f32
}

__device__ __forceinline__ float fast_tanh(float x) {
#if __has_builtin(__builtin_amdgcn_tanhf)
  return __builtin_amdgcn_tanhf(x);                    // v_tanh_f32
#else
  // branchless: tanh(x) = 1 - 2/(2^(2x*log2e)+1); saturates correctly at +-inf
  float e = __builtin_amdgcn_exp2f((2.0f * LOG2E_) * x);
  return 1.0f - 2.0f * __builtin_amdgcn_rcpf(e + 1.0f);
#endif
}

__device__ __forceinline__ v8f wmma_f16(v8f acc, v16h a, v16h b) {
  return __builtin_amdgcn_wmma_f32_16x16x32_f16(
      false, a, false, b, (short)0, acc, false, false);
}

// B-matrix fragment (32x16 f16): lane L holds N = n0+(L&15);
// halves e=0..15 hold K = kbase + (L<16 ? e : 16+e).  B[k][n] = W[n*ldw + k].
__device__ __forceinline__ void load_bfrag(v16h& frag, const float* __restrict__ W,
                                           int ldw, int n0, int kbase, int lane) {
  _Float16* p = (_Float16*)&frag;
  const int N    = n0 + (lane & 15);
  const int kofs = kbase + ((lane < 16) ? 0 : 16);
#pragma unroll
  for (int e = 0; e < 16; ++e)
    p[e] = (_Float16)W[N * ldw + kofs + e];
}

// Fused decoder recurrent weight: Wf[k][n] = dWhh[n*H+k] + sum_j dWih[n*F+j]*Wo[j*H+k]
__device__ __forceinline__ void load_fused_frag(v16h& frag,
    const float* __restrict__ dWih, const float* __restrict__ dWhh,
    const float* __restrict__ Wo, int n0, int kbase, int lane) {
  _Float16* p = (_Float16*)&frag;
  const int N    = n0 + (lane & 15);
  const int kofs = kbase + ((lane < 16) ? 0 : 16);
#pragma unroll
  for (int e = 0; e < 16; ++e) {
    const int k = kofs + e;
    float a = dWhh[N * H_ + k];
    for (int j = 0; j < F_; ++j)
      a += dWih[N * F_ + j] * Wo[j * H_ + k];
    p[e] = (_Float16)a;
  }
}

// Scatter one f16 into the LDS A-matrix (16x32 per k-tile) layout.
__device__ __forceinline__ void store_a_half(_Float16* abuf, int col, int row, _Float16 v) {
  const int kk = col >> 5;
  const int k  = col & 31;
  const int L  = row + ((k >> 4) << 4);
  const int vv = (k & 15) >> 1;
  abuf[(kk * 32 + L) * 16 + vv * 2 + (k & 1)] = v;
}

// ====================== pre-GEMM: zx[t][b][512] = x_t @ eWih^T ======================
// Fully parallel: one block per (t, batch-tile); 16384 blocks use the whole chip.
// t is reversed so early timesteps are written LAST -> hot in the 192MB L2 when
// the recurrent kernel starts consuming at t=0.
__global__ void __launch_bounds__(NT_)
zx_pregemm(const float* __restrict__ x, const float* __restrict__ eWih,
           float* __restrict__ zx)
{
  __shared__ __align__(32) _Float16 xa[2 * 32 * 16];
  const int tid = threadIdx.x, lane = tid & 31, wave = tid >> 5;
  const int t = (T_ - 1) - (blockIdx.x >> 3), mb = (blockIdx.x & 7) * 16;
  const int n0 = wave * 32;

  v16h wx[2][2];
#pragma unroll
  for (int nt = 0; nt < 2; ++nt)
#pragma unroll
    for (int kt = 0; kt < 2; ++kt)
      load_bfrag(wx[kt][nt], eWih, F_, n0 + 16 * nt, 32 * kt, lane);

#pragma unroll
  for (int jj = 0; jj < 2; ++jj) {
    int e = tid + NT_ * jj;
    int row = e >> 6, col = e & 63;
    float v = x[((size_t)(mb + row) * T_ + t) * F_ + col];
    store_a_half(xa, col, row, (_Float16)v);
  }
  __syncthreads();

  v16h ax[2];
#pragma unroll
  for (int kt = 0; kt < 2; ++kt) ax[kt] = *(const v16h*)&xa[(kt * 32 + lane) * 16];

  float* dst = zx + ((size_t)t * B_ + mb) * G4H_;
  const int m0 = (lane < 16) ? 0 : 8, nn = lane & 15;
#pragma unroll
  for (int nt = 0; nt < 2; ++nt) {
    v8f acc = {};
#pragma unroll
    for (int kt = 0; kt < 2; ++kt) acc = wmma_f16(acc, ax[kt], wx[kt][nt]);
    const int nb = n0 + 16 * nt;
#pragma unroll
    for (int r = 0; r < 8; ++r)
      __builtin_nontemporal_store(acc[r], &dst[(size_t)(m0 + r) * G4H_ + nb + nn]);
  }
}

// ====================== recurrent kernel ======================
__global__ void __launch_bounds__(NT_)
lstm_ed_kernel(const float* __restrict__ x,
               const float* __restrict__ eWih, const float* __restrict__ eWhh,
               const float* __restrict__ ebih, const float* __restrict__ ebhh,
               const float* __restrict__ dWih, const float* __restrict__ dWhh,
               const float* __restrict__ dbih, const float* __restrict__ dbhh,
               const float* __restrict__ Wo,   const float* __restrict__ bo,
               const float* __restrict__ zx,   int use_zx,
               float* __restrict__ out)
{
  __shared__ __align__(32) _Float16 ha[4 * 32 * 16];  // h in A layout (K=128)
  __shared__ __align__(32) _Float16 xa[2 * 32 * 16];  // x in A layout (fallback only)
  __shared__ float zbuf[16][G4H_];
  __shared__ float obuf[16][F_];

  const int tid  = threadIdx.x;
  const int lane = tid & 31;
  const int wave = tid >> 5;
  const int mb   = blockIdx.x * 16;
  const int n0   = wave * 32;
  const int m0   = (lane < 16) ? 0 : 8;
  const int nn   = lane & 15;

  v16h wx[2][2];   // encoder Wih^T fragments (fallback path only)
  v16h wh[4][2];   // recurrent weight fragments (enc: Whh^T, dec: fused)
  v16h wof[4];     // Wo^T fragments (decoder projection, waves 0..3)

  // ---------------- encoder weight fragments ----------------
#pragma unroll
  for (int nt = 0; nt < 2; ++nt) {
#pragma unroll
    for (int kt = 0; kt < 4; ++kt)
      load_bfrag(wh[kt][nt], eWhh, H_, n0 + 16 * nt, 32 * kt, lane);
    if (!use_zx) {
#pragma unroll
      for (int kt = 0; kt < 2; ++kt)
        load_bfrag(wx[kt][nt], eWih, F_, n0 + 16 * nt, 32 * kt, lane);
    }
  }

  float bi[4], bf[4], bg[4], bo4[4];
#pragma unroll
  for (int j = 0; j < 4; ++j) {
    int col = (tid + NT_ * j) & (H_ - 1);
    bi[j]  = ebih[col]          + ebhh[col];
    bf[j]  = ebih[H_ + col]     + ebhh[H_ + col];
    bg[j]  = ebih[2 * H_ + col] + ebhh[2 * H_ + col];
    bo4[j] = ebih[3 * H_ + col] + ebhh[3 * H_ + col];
  }
  float cst[4] = {0.f, 0.f, 0.f, 0.f};

  for (int e = tid; e < 4 * 32 * 16; e += NT_) ha[e] = (_Float16)0.f;
  if (!use_zx) {
#pragma unroll
    for (int jj = 0; jj < 2; ++jj) {
      int e = tid + NT_ * jj;
      int row = e >> 6, col = e & 63;
      store_a_half(xa, col, row, (_Float16)x[((size_t)(mb + row) * T_) * F_ + col]);
    }
  } else {
    // warm up: prefetch the first zx tile (32KB: 512 threads x 64B)
    __builtin_prefetch((const char*)(zx + (size_t)mb * G4H_) + tid * 64, 0, 1);
  }
  __syncthreads();

  // ======================= encoder =======================
  for (int t = 0; t < T_; ++t) {
    // phase A: z = zx_t + h*Whh^T   (or + x_t*Wih^T inline in fallback)
    v16h ahf[4];
#pragma unroll
    for (int kt = 0; kt < 4; ++kt) ahf[kt] = *(const v16h*)&ha[(kt * 32 + lane) * 16];
    v16h axf[2] = {};
    if (!use_zx) {
#pragma unroll
      for (int kt = 0; kt < 2; ++kt) axf[kt] = *(const v16h*)&xa[(kt * 32 + lane) * 16];
    } else if (t + 1 < T_) {
      // cover next step's whole 32KB zx tile one step ahead
      const char* pf = (const char*)(zx + ((size_t)(t + 1) * B_ + mb) * G4H_);
      __builtin_prefetch(pf + tid * 64, 0, 1);
    }
#pragma unroll
    for (int nt = 0; nt < 2; ++nt) {
      const int nb = n0 + 16 * nt;
      v8f acc = {};
      if (use_zx) {
        const float* zrow = zx + ((size_t)t * B_ + mb) * G4H_ + nb + nn;
#pragma unroll
        for (int r = 0; r < 8; ++r)
          acc[r] = __builtin_nontemporal_load(&zrow[(size_t)(m0 + r) * G4H_]);
      } else {
#pragma unroll
        for (int kt = 0; kt < 2; ++kt) acc = wmma_f16(acc, axf[kt], wx[kt][nt]);
      }
#pragma unroll
      for (int kt = 0; kt < 4; ++kt) acc = wmma_f16(acc, ahf[kt], wh[kt][nt]);
#pragma unroll
      for (int r = 0; r < 8; ++r) zbuf[m0 + r][nb + nn] = acc[r];
    }
    __syncthreads();

    // phase B: gates -> new h (and stage next x in fallback mode)
#pragma unroll
    for (int j = 0; j < 4; ++j) {
      int e = tid + NT_ * j;
      int row = e >> 7, col = e & (H_ - 1);
      float iv = fast_sigmoid(zbuf[row][col]          + bi[j]);
      float fv = fast_sigmoid(zbuf[row][H_ + col]     + bf[j]);
      float gv = fast_tanh   (zbuf[row][2 * H_ + col] + bg[j]);
      float ov = fast_sigmoid(zbuf[row][3 * H_ + col] + bo4[j]);
      float cn = fv * cst[j] + iv * gv;
      cst[j] = cn;
      store_a_half(ha, col, row, (_Float16)(ov * fast_tanh(cn)));
    }
    if (!use_zx && t + 1 < T_) {
#pragma unroll
      for (int jj = 0; jj < 2; ++jj) {
        int e = tid + NT_ * jj;
        int row = e >> 6, col = e & 63;
        const float* src = &x[((size_t)(mb + row) * T_ + t + 1) * F_ + col];
        if (t + 5 < T_) __builtin_prefetch(src + 4 * F_, 0, 1);
        store_a_half(xa, col, row, (_Float16)(*src));
      }
    }
    __syncthreads();
  }

  // ---------------- decoder: fused recurrent weight + projection ----------------
#pragma unroll
  for (int nt = 0; nt < 2; ++nt)
#pragma unroll
    for (int kt = 0; kt < 4; ++kt)
      load_fused_frag(wh[kt][nt], dWih, dWhh, Wo, n0 + 16 * nt, 32 * kt, lane);
  if (wave < 4) {
#pragma unroll
    for (int kt = 0; kt < 4; ++kt)
      load_bfrag(wof[kt], Wo, H_, wave * 16, 32 * kt, lane);
  }
  // fused bias: b + (bo @ dWih^T)
#pragma unroll
  for (int j = 0; j < 4; ++j) {
    int col = (tid + NT_ * j) & (H_ - 1);
    float zb[4];
#pragma unroll
    for (int g = 0; g < 4; ++g) {
      const int n = g * H_ + col;
      float a = 0.f;
      for (int jj = 0; jj < F_; ++jj) a += bo[jj] * dWih[n * F_ + jj];
      zb[g] = a;
    }
    bi[j]  = dbih[col]          + dbhh[col]          + zb[0];
    bf[j]  = dbih[H_ + col]     + dbhh[H_ + col]     + zb[1];
    bg[j]  = dbih[2 * H_ + col] + dbhh[2 * H_ + col] + zb[2];
    bo4[j] = dbih[3 * H_ + col] + dbhh[3 * H_ + col] + zb[3];
  }
  float boc[2];
#pragma unroll
  for (int jj = 0; jj < 2; ++jj) boc[jj] = bo[(tid + NT_ * jj) & (F_ - 1)];

  // ======================= decoder =======================
  // Step s: out_t = h@Wo^T + bo at time T-1-s;  z = h@Wfused + fused_bias.
  for (int s = 0; s < T_; ++s) {
    // phase A: both matmuls read only h
    v16h ahf[4];
#pragma unroll
    for (int kt = 0; kt < 4; ++kt) ahf[kt] = *(const v16h*)&ha[(kt * 32 + lane) * 16];
#pragma unroll
    for (int nt = 0; nt < 2; ++nt) {
      v8f acc = {};
#pragma unroll
      for (int kt = 0; kt < 4; ++kt) acc = wmma_f16(acc, ahf[kt], wh[kt][nt]);
      const int nb = n0 + 16 * nt;
#pragma unroll
      for (int r = 0; r < 8; ++r) zbuf[m0 + r][nb + nn] = acc[r];
    }
    if (wave < 4) {
      v8f pacc = {};
#pragma unroll
      for (int kt = 0; kt < 4; ++kt) pacc = wmma_f16(pacc, ahf[kt], wof[kt]);
      const int nb = wave * 16;
#pragma unroll
      for (int r = 0; r < 8; ++r) obuf[m0 + r][nb + nn] = pacc[r];
    }
    __syncthreads();

    // phase B: emit out_t (reversed time, streaming store) + gates -> new h
#pragma unroll
    for (int jj = 0; jj < 2; ++jj) {
      int e = tid + NT_ * jj;
      int row = e >> 6, col = e & 63;
      __builtin_nontemporal_store(
          obuf[row][col] + boc[jj],
          &out[((size_t)(mb + row) * T_ + (T_ - 1 - s)) * F_ + col]);
    }
#pragma unroll
    for (int j = 0; j < 4; ++j) {
      int e = tid + NT_ * j;
      int row = e >> 7, col = e & (H_ - 1);
      float iv = fast_sigmoid(zbuf[row][col]          + bi[j]);
      float fv = fast_sigmoid(zbuf[row][H_ + col]     + bf[j]);
      float gv = fast_tanh   (zbuf[row][2 * H_ + col] + bg[j]);
      float ov = fast_sigmoid(zbuf[row][3 * H_ + col] + bo4[j]);
      float cn = fv * cst[j] + iv * gv;
      cst[j] = cn;
      store_a_half(ha, col, row, (_Float16)(ov * fast_tanh(cn)));
    }
    __syncthreads();
  }
}

extern "C" void kernel_launch(void* const* d_in, const int* in_sizes, int n_in,
                              void* d_out, int out_size, void* d_ws, size_t ws_size,
                              hipStream_t stream) {
  const float* xp   = (const float*)d_in[0];
  const float* eWih = (const float*)d_in[1];
  const float* eWhh = (const float*)d_in[2];
  const float* ebih = (const float*)d_in[3];
  const float* ebhh = (const float*)d_in[4];
  const float* dWih = (const float*)d_in[5];
  const float* dWhh = (const float*)d_in[6];
  const float* dbih = (const float*)d_in[7];
  const float* dbhh = (const float*)d_in[8];
  const float* Wop  = (const float*)d_in[9];
  const float* bop  = (const float*)d_in[10];
  float* outp       = (float*)d_out;

  const size_t zx_bytes = (size_t)T_ * B_ * G4H_ * sizeof(float);  // 512 MB
  const int use_zx = (ws_size >= zx_bytes) ? 1 : 0;
  float* zxp = (float*)d_ws;

  if (use_zx) {
    // whole-chip parallel pre-GEMM of the non-recurrent encoder term
    zx_pregemm<<<dim3(T_ * (B_ / 16)), dim3(NT_), 0, stream>>>(xp, eWih, zxp);
  }
  lstm_ed_kernel<<<dim3(B_ / 16), dim3(NT_), 0, stream>>>(
      xp, eWih, eWhh, ebih, ebhh, dWih, dWhh, dbih, dbhh, Wop, bop,
      zxp, use_zx, outp);
}
